// JanossyPoolingImproper_1408749273396
// MI455X (gfx1250) — matrix operand order repaired and checked by reading, not compiled, not run on hardware
//
#include <hip/hip_runtime.h>
#include <hip/hip_bf16.h>

// ---------------------------------------------------------------------------
// Janossy pooling (3 perms of 4 gathered neighbors) -> shared 3x[512x512] MLP
// with ReLU -> sum over perms -> 512x2 projection.
// Compute-bound (~472 GFLOP vs ~32MB HBM): run matmuls on v_wmma_f32_16x16x32_f16
// (f16 inputs, f32 accumulate). Activations staged in LDS (double buffered),
// weights pre-converted once to f16 transposed layout in d_ws so each lane's
// B fragment is one contiguous 32B load (L2-resident).
// Loop nest: single K loop, A fragments loaded once per k-step and reused by
// 4 column tiles -> 12 independent WMMAs per k-step (96 VGPRs of accumulators).
// ---------------------------------------------------------------------------

typedef __attribute__((ext_vector_type(16))) _Float16 v16h;
typedef __attribute__((ext_vector_type(8)))  _Float16 v8h;
typedef __attribute__((ext_vector_type(4)))  _Float16 v4h;
typedef __attribute__((ext_vector_type(8)))  float    v8f;
typedef __attribute__((ext_vector_type(4)))  float    v4f;

#define HD      512            // hidden dim == 4*FEAT
#define FEAT    128            // per-node feature dim
#define ITEMS   16             // items (original rows) per block
#define ROWS    48             // 3 perms * ITEMS activation rows
#define LSTRIDE 520            // f16 per LDS row: 512 + 8 pad (dword stride 260 ≡ 4 mod 64)

__device__ __constant__ int PERM[3][4] = {{0,1,2,3},{2,1,3,0},{3,1,0,2}};

union AF { v16h v; v8h h[2]; };

#define WMMA_F16(a, b, c) \
  __builtin_amdgcn_wmma_f32_16x16x32_f16(false, (a), false, (b), (short)0, (c), false, false)

// Convert W1/W2/W3 (f32, row-major [k][n]) -> wt (f16, [layer][n][k]).
__global__ void wconvert_kernel(const float* __restrict__ W1,
                                const float* __restrict__ W2,
                                const float* __restrict__ W3,
                                _Float16* __restrict__ wt) {
  int e = blockIdx.x * blockDim.x + threadIdx.x;   // 0 .. 3*512*512-1
  int L = e >> 18;                                  // / (512*512)
  int r = e & 262143;
  int n = r >> 9;
  int k = r & 511;
  const float* W = (L == 0) ? W1 : (L == 1) ? W2 : W3;
  wt[e] = (_Float16)W[k * HD + n];
}

__global__ __launch_bounds__(256)
void janossy_kernel(const float* __restrict__ htab,
                    const int* __restrict__ idx,
                    const _Float16* __restrict__ wt,
                    const float* __restrict__ b1,
                    const float* __restrict__ b2,
                    const float* __restrict__ b3,
                    const float* __restrict__ Wo,
                    const float* __restrict__ bo,
                    float* __restrict__ out,
                    int Nimp) {
  extern __shared__ _Float16 smem[];
  _Float16* Xin  = smem;                  // [ROWS][LSTRIDE]
  _Float16* Xout = smem + ROWS * LSTRIDE; // [ROWS][LSTRIDE]

  const int tid      = threadIdx.x;
  const int lane     = tid & 31;
  const int wave     = tid >> 5;
  const int laneMod  = lane & 15;
  const int laneHalf = lane >> 4;
  const int n0       = blockIdx.x * ITEMS;

  // ---- Stage 0: gather h[idx[perm]] into X f16 rows [p*16+i][512] ----------
  for (int u = tid; u < ROWS * (HD / 4); u += 256) {
    int row = u >> 7;                 // / (HD/4 = 128)
    int q   = u & 127;
    int seg = q >> 5;                 // which of 4 neighbor slots
    int off = (q & 31) << 2;          // float4 offset within 128
    int p   = row >> 4;               // perm
    int i   = row & 15;               // item within block
    int gi  = n0 + i; if (gi >= Nimp) gi = Nimp - 1;
    int node = idx[gi * 4 + PERM[p][seg]];
    v4f s = *(const v4f*)(htab + (size_t)node * FEAT + off);
    v4h d; d.x = (_Float16)s.x; d.y = (_Float16)s.y;
           d.z = (_Float16)s.z; d.w = (_Float16)s.w;
    *(v4h*)(Xin + row * LSTRIDE + seg * FEAT + off) = d;
  }
  __syncthreads();

  // ---- Stages 1..3: X = relu(X @ W + b), via WMMA f32_16x16x32_f16 ----------
  // Per wave: 4 column tiles (col = wave*64 + j*16 + laneMod), 3 row tiles
  // (one per permutation). 12 accumulators live across the whole K loop.
  for (int layer = 0; layer < 3; ++layer) {
    const _Float16* Wl  = wt + (size_t)layer * HD * HD;  // [n][k] f16
    const float* bias   = (layer == 0) ? b1 : (layer == 1) ? b2 : b3;

    const int colBase = wave * 64 + laneMod;
    // B fragment: lane<16 -> column col, K k0..k0+15 ; lane>=16 -> K k0+16..k0+31
    const _Float16* wbase = Wl + (size_t)colBase * HD + laneHalf * 16;
    // A fragment: lane<16 -> row laneMod, K k0..k0+15 ; lane>=16 -> K k0+16..k0+31
    const _Float16* arow  = Xin + laneMod * LSTRIDE + laneHalf * 16;

    v8f acc[4][3];
    #pragma unroll
    for (int j = 0; j < 4; ++j)
      #pragma unroll
      for (int m = 0; m < 3; ++m)
        acc[j][m] = (v8f){0.f, 0.f, 0.f, 0.f, 0.f, 0.f, 0.f, 0.f};

    #pragma unroll 1
    for (int ks = 0; ks < HD / 32; ++ks) {
      const int k0 = ks * 32;
      const _Float16* ap = arow + k0;
      AF a0; a0.h[0] = *(const v8h*)(ap);
             a0.h[1] = *(const v8h*)(ap + 8);
      AF a1; a1.h[0] = *(const v8h*)(ap + 16 * LSTRIDE);
             a1.h[1] = *(const v8h*)(ap + 16 * LSTRIDE + 8);
      AF a2; a2.h[0] = *(const v8h*)(ap + 32 * LSTRIDE);
             a2.h[1] = *(const v8h*)(ap + 32 * LSTRIDE + 8);
      #pragma unroll
      for (int j = 0; j < 4; ++j) {
        AF b; b.v = *(const v16h*)(wbase + (size_t)j * 16 * HD + k0);
        acc[j][0] = WMMA_F16(a0.v, b.v, acc[j][0]);
        acc[j][1] = WMMA_F16(a1.v, b.v, acc[j][1]);
        acc[j][2] = WMMA_F16(a2.v, b.v, acc[j][2]);
      }
    }

    // Epilogue: bias + ReLU, f32 -> f16, scatter per C/D layout
    // (VGPR r holds M=r for lanes 0-15, M=r+8 for lanes 16-31)
    #pragma unroll
    for (int j = 0; j < 4; ++j) {
      const int col = colBase + j * 16;
      const float bv = bias[col];
      #pragma unroll
      for (int r = 0; r < 8; ++r) {
        const int rr = laneHalf * 8 + r;
        Xout[(rr)      * LSTRIDE + col] = (_Float16)fmaxf(acc[j][0][r] + bv, 0.0f);
        Xout[(16 + rr) * LSTRIDE + col] = (_Float16)fmaxf(acc[j][1][r] + bv, 0.0f);
        Xout[(32 + rr) * LSTRIDE + col] = (_Float16)fmaxf(acc[j][2][r] + bv, 0.0f);
      }
    }
    __syncthreads();
    _Float16* t = Xin; Xin = Xout; Xout = t;
  }

  // ---- Stage 4: Janossy sum over perms + 512x2 projection ------------------
  // 32 (item,channel) pairs, 8 lanes each; reduce within groups of 8.
  {
    const _Float16* A = Xin;          // after 3 swaps, final activations
    const int g = tid >> 3;           // 0..31
    const int l = tid & 7;
    const int i = g >> 1;
    const int c = g & 1;
    float acc = 0.0f;
    for (int k = l; k < HD; k += 8) {
      float s = (float)A[i * LSTRIDE + k]
              + (float)A[(16 + i) * LSTRIDE + k]
              + (float)A[(32 + i) * LSTRIDE + k];
      acc += s * Wo[k * 2 + c];
    }
    acc += __shfl_down(acc, 4, 8);
    acc += __shfl_down(acc, 2, 8);
    acc += __shfl_down(acc, 1, 8);
    if (l == 0 && (n0 + i) < Nimp)
      out[(size_t)(n0 + i) * 2 + c] = acc + bo[c];
  }
}

extern "C" void kernel_launch(void* const* d_in, const int* in_sizes, int n_in,
                              void* d_out, int out_size, void* d_ws, size_t ws_size,
                              hipStream_t stream) {
  const float* h  = (const float*)d_in[0];
  const int*   ix = (const int*)  d_in[1];
  const float* W1 = (const float*)d_in[2];
  const float* b1 = (const float*)d_in[3];
  const float* W2 = (const float*)d_in[4];
  const float* b2 = (const float*)d_in[5];
  const float* W3 = (const float*)d_in[6];
  const float* b3 = (const float*)d_in[7];
  const float* Wo = (const float*)d_in[8];
  const float* bo = (const float*)d_in[9];
  float* out = (float*)d_out;

  _Float16* wt = (_Float16*)d_ws;          // 3 * 512 * 512 * 2B = 1.5 MB scratch
  const int Nimp = in_sizes[1] / 4;

  // One-time-per-call weight convert (deterministic; graph-capture safe).
  wconvert_kernel<<<(3 * HD * HD) / 256, 256, 0, stream>>>(W1, W2, W3, wt);

  const int blocks = (Nimp + ITEMS - 1) / ITEMS;
  const size_t shmem = (size_t)2 * ROWS * LSTRIDE * sizeof(_Float16);  // ~97.5 KB
  janossy_kernel<<<blocks, 256, shmem, stream>>>(h, ix, wt, b1, b2, b3,
                                                 Wo, bo, out, Nimp);
}
